// Attention_52166672777669
// MI455X (gfx1250) — matrix operand order repaired and checked by reading, not compiled, not run on hardware
//
#include <hip/hip_runtime.h>

// ---------------------------------------------------------------------------
// Types matching gfx1250 WMMA operand shapes
// ---------------------------------------------------------------------------
typedef __attribute__((ext_vector_type(16))) _Float16 v16h;
typedef __attribute__((ext_vector_type(8)))  _Float16 v8h;
typedef __attribute__((ext_vector_type(8)))  float    v8f;

#define CD  512   // channel dim C
#define BQN 256   // Bq
#define BKN 256   // Bk
#define NKN 512   // Nk
#define LDS_LD 520  // padded row stride (halves) for staged tiles

static __device__ __forceinline__ v16h cat16(v8h lo, v8h hi) {
  v16h a;
#pragma unroll
  for (int i = 0; i < 8; ++i) { a[i] = lo[i]; a[8 + i] = hi[i]; }
  return a;
}

// A-fragment (16x32 f16, M x K) from a row-major tile.
// ISA layout: lane m = l%16, half h = l/16; elems 0..7 -> K = h*8..h*8+7,
// elems 8..15 -> K = 16+h*8 .. 16+h*8+7  => two contiguous 16B loads.
static __device__ __forceinline__ v16h load_a_frag(const _Float16* base, int ld,
                                                   int row0, int k0, int lane) {
  int m = lane & 15, h = lane >> 4;
  const _Float16* p = base + (row0 + m) * ld + k0 + h * 8;
  return cat16(*(const v8h*)p, *(const v8h*)(p + 16));
}

// B-fragment (32x16 f16, K x N): lane l holds column N=l%16; elems 0..7 -> K = h*8..h*8+7,
// elems 8..15 -> K = 16+h*8 .. 16+h*8+7 (strided by ld).
static __device__ __forceinline__ v16h load_b_frag(const _Float16* p, int ld, int lane) {
  int h = lane >> 4;
  v16h b;
#pragma unroll
  for (int s = 0; s < 8; ++s) { b[s] = p[(h * 8 + s) * ld]; b[8 + s] = p[(16 + h * 8 + s) * ld]; }
  return b;
}

static __device__ __forceinline__ v8f wmma_f16(v16h a, v16h b, v8f c) {
  return __builtin_amdgcn_wmma_f32_16x16x32_f16(false, a, false, b,
                                                (short)0, c, false, false);
}

// ---------------------------------------------------------------------------
// Kernel 1: W[j][c] -> WT[c][j] so GEMM B-fragments are contiguous per lane.
// ---------------------------------------------------------------------------
__global__ void transpose512_kernel(const _Float16* __restrict__ W,
                                    _Float16* __restrict__ WT) {
  int idx = blockIdx.x * 256 + threadIdx.x;   // 0 .. 512*512-1
  int c = idx >> 9, j = idx & 511;
  WT[idx] = W[j * CD + c];                    // WT[c*512 + j]
}

// ---------------------------------------------------------------------------
// Kernel 2: fused LayerNorm + (LN(x) @ W^T) with WMMA.
// 32 rows / workgroup, 256 threads (8 waves).
// Wave w: row tile w&1 (16 rows), column quarter w>>1 (8 tiles of 16 cols)
//   -> 8 v8f accumulators (64 VGPRs), no spills.
// transposeOut: store K per-batch transposed  KT[row/512][j][row%512].
// ---------------------------------------------------------------------------
__global__ void __launch_bounds__(256, 1)
ln_proj_kernel(const _Float16* __restrict__ X,
               const _Float16* __restrict__ gamma,
               const _Float16* __restrict__ beta,
               const _Float16* __restrict__ WT,
               _Float16* __restrict__ out,
               int transposeOut) {
  extern __shared__ char smem[];
  _Float16* Xs = (_Float16*)smem;             // [32][LDS_LD]
  const int tid = threadIdx.x, wave = tid >> 5, lane = tid & 31;
  const int row0 = blockIdx.x * 32;

  // per-lane gamma/beta (16 contiguous channels per lane)
  v8h g0 = *(const v8h*)(gamma + lane * 16);
  v8h g1 = *(const v8h*)(gamma + lane * 16 + 8);
  v8h b0 = *(const v8h*)(beta  + lane * 16);
  v8h b1 = *(const v8h*)(beta  + lane * 16 + 8);

  // --- LayerNorm: wave w handles local rows [w*4, w*4+4) ---
#pragma unroll
  for (int i = 0; i < 4; ++i) {
    int lr = wave * 4 + i;
    const _Float16* xr = X + (size_t)(row0 + lr) * CD + lane * 16;
    v8h x0 = *(const v8h*)xr;
    v8h x1 = *(const v8h*)(xr + 8);
    float s = 0.f, ss = 0.f;
#pragma unroll
    for (int e = 0; e < 8; ++e) {
      float a = (float)x0[e], b = (float)x1[e];
      s += a + b; ss += a * a + b * b;
    }
#pragma unroll
    for (int m = 1; m < 32; m <<= 1) {
      s  += __shfl_xor(s,  m, 32);
      ss += __shfl_xor(ss, m, 32);
    }
    float mean = s * (1.f / 512.f);
    float var  = ss * (1.f / 512.f) - mean * mean;
    float rstd = rsqrtf(var + 1e-5f);
    v8h y0, y1;
#pragma unroll
    for (int e = 0; e < 8; ++e) {
      y0[e] = (_Float16)(((float)x0[e] - mean) * rstd * (float)g0[e] + (float)b0[e]);
      y1[e] = (_Float16)(((float)x1[e] - mean) * rstd * (float)g1[e] + (float)b1[e]);
    }
    _Float16* dst = Xs + lr * LDS_LD + lane * 16;
    *(v8h*)dst = y0;
    *(v8h*)(dst + 8) = y1;
  }
  __syncthreads();

  // --- GEMM: out[row][j] = sum_c LN[row][c] * WT[c][j] ---
  const int rt = wave & 1;            // row tile (16 rows)
  const int jbase = (wave >> 1) * 128; // column quarter
  v8f zero = {};
  v8f acc[8];
#pragma unroll
  for (int t = 0; t < 8; ++t) acc[t] = zero;

  for (int c0 = 0; c0 < CD; c0 += 32) {
    v16h a = load_a_frag(Xs, LDS_LD, rt * 16, c0, lane);
    const _Float16* bp = WT + (size_t)c0 * CD + jbase + (lane & 15);
    v16h b[8];
#pragma unroll
    for (int t = 0; t < 8; ++t) b[t] = load_b_frag(bp + t * 16, CD, lane);
#pragma unroll
    for (int t = 0; t < 8; ++t) acc[t] = wmma_f16(a, b[t], acc[t]);
  }

  // store (C/D layout: lane l, vgpr r -> M = r + 8*(l/16), N = l%16)
  const int h = lane >> 4, n = lane & 15;
#pragma unroll
  for (int t = 0; t < 8; ++t) {
    int j0 = jbase + t * 16;
#pragma unroll
    for (int r = 0; r < 8; ++r) {
      int row = row0 + rt * 16 + r + 8 * h;
      int j = j0 + n;
      _Float16 v = (_Float16)acc[t][r];
      if (!transposeOut)
        out[(size_t)row * CD + j] = v;
      else  // KT[kb][j][n] : kb = row/512, n = row%512
        out[((size_t)(row >> 9)) * (CD * NKN) + (size_t)j * NKN + (row & 511)] = v;
    }
  }
}

// ---------------------------------------------------------------------------
// Kernel 3: attention.  grid = (k=256, qtile=8), 256 threads (8 waves).
//  S = Q @ K_k^T  (scale + pad mask) -> softmax (registers + shfl + tiny LDS)
//  X = P @ V      (V = kx[k]),  out[q][k][c]
// Wave w: row tile w&1 (16 q rows), n/c quarter w>>1 -> 8 v8f accumulators.
// ---------------------------------------------------------------------------
__global__ void __launch_bounds__(256, 1)
attn_kernel(const _Float16* __restrict__ Q,
            const _Float16* __restrict__ KT,
            const _Float16* __restrict__ V,
            const unsigned char* __restrict__ mask,
            _Float16* __restrict__ out) {
  extern __shared__ char smem[];
  _Float16* Qs = (_Float16*)smem;                      // [32][LDS_LD], reused as P
  float* red = (float*)(smem + 32 * LDS_LD * 2);       // [32][8] {max x4, sum x4}
  const int tid = threadIdx.x, wave = tid >> 5, lane = tid & 31;
  const int k = blockIdx.x;
  const int q0 = blockIdx.y * 32;

  // --- stage Q tile (32x512 f16) into LDS with wide b128 copies ---
  {
    int lr = tid >> 3;           // 0..31
    int cb = (tid & 7) * 64;
    const _Float16* src = Q + (size_t)(q0 + lr) * CD + cb;
    _Float16* dst = Qs + lr * LDS_LD + cb;
#pragma unroll
    for (int i = 0; i < 8; ++i)
      *(v8h*)(dst + i * 8) = *(const v8h*)(src + i * 8);
  }
  __syncthreads();

  const int rt = wave & 1;          // row tile
  const int quarter = wave >> 1;    // n / c quarter
  const int nbase = quarter * 128;
  const int h = lane >> 4, nn = lane & 15;

  v8f zero = {};
  v8f acc[8];
#pragma unroll
  for (int t = 0; t < 8; ++t) acc[t] = zero;

  // --- S = Q @ K^T, full S slice kept in registers ---
  const _Float16* ktb = KT + (size_t)k * (CD * NKN);
  for (int c0 = 0; c0 < CD; c0 += 32) {
    v16h a = load_a_frag(Qs, LDS_LD, rt * 16, c0, lane);
    const _Float16* bp = ktb + (size_t)c0 * NKN + nbase + (lane & 15);
    v16h b[8];
#pragma unroll
    for (int t = 0; t < 8; ++t) b[t] = load_b_frag(bp + t * 16, NKN, lane);
#pragma unroll
    for (int t = 0; t < 8; ++t) acc[t] = wmma_f16(a, b[t], acc[t]);
  }

  // --- scale + key-padding mask (masked -> -1e30 so exp() -> 0) ---
  const float scale = 0.044194173824159216f;  // 512^-0.5
#pragma unroll
  for (int t = 0; t < 8; ++t) {
    int n = nbase + t * 16 + nn;
    bool pad = mask[k * NKN + n] != 0;
#pragma unroll
    for (int r = 0; r < 8; ++r)
      acc[t][r] = pad ? -1e30f : acc[t][r] * scale;
  }

  // --- softmax over n (512): shfl within 16-lane halves + LDS across 4 waves
  float rowmax[8], rowsum[8];
#pragma unroll
  for (int r = 0; r < 8; ++r) {
    float mx = -3.0e38f;
#pragma unroll
    for (int t = 0; t < 8; ++t) mx = fmaxf(mx, acc[t][r]);
    for (int m = 1; m < 16; m <<= 1) mx = fmaxf(mx, __shfl_xor(mx, m, 32));
    if (nn == 0) red[(rt * 16 + r + 8 * h) * 8 + quarter] = mx;
  }
  __syncthreads();
#pragma unroll
  for (int r = 0; r < 8; ++r) {
    int lr = rt * 16 + r + 8 * h;
    float m01 = fmaxf(red[lr * 8 + 0], red[lr * 8 + 1]);
    float m23 = fmaxf(red[lr * 8 + 2], red[lr * 8 + 3]);
    rowmax[r] = fmaxf(m01, m23);
  }
#pragma unroll
  for (int r = 0; r < 8; ++r) {
    float s = 0.f;
#pragma unroll
    for (int t = 0; t < 8; ++t) {
      float e = __expf(acc[t][r] - rowmax[r]);
      acc[t][r] = e; s += e;
    }
    for (int m = 1; m < 16; m <<= 1) s += __shfl_xor(s, m, 32);
    if (nn == 0) red[(rt * 16 + r + 8 * h) * 8 + 4 + quarter] = s;
  }
  __syncthreads();
#pragma unroll
  for (int r = 0; r < 8; ++r) {
    int lr = rt * 16 + r + 8 * h;
    rowsum[r] = (red[lr * 8 + 4] + red[lr * 8 + 5]) +
                (red[lr * 8 + 6] + red[lr * 8 + 7]);
  }

  // --- write P (f16) into the Q buffer (all waves past Q reads by now) ---
#pragma unroll
  for (int r = 0; r < 8; ++r) {
    int lr = rt * 16 + r + 8 * h;
    float inv = 1.f / rowsum[r];
#pragma unroll
    for (int t = 0; t < 8; ++t)
      Qs[lr * LDS_LD + nbase + t * 16 + nn] = (_Float16)(acc[t][r] * inv);
  }
  __syncthreads();

  // --- X = P @ V,  V = kx[k]  (B fragments contiguous per lane) ---
  const int cbase = quarter * 128;
  v8f acc2[8];
#pragma unroll
  for (int t = 0; t < 8; ++t) acc2[t] = zero;
  const _Float16* vb = V + (size_t)k * NKN * CD;
  for (int n0 = 0; n0 < NKN; n0 += 32) {
    v16h a = load_a_frag(Qs, LDS_LD, rt * 16, n0, lane);
    const _Float16* bp = vb + (size_t)n0 * CD + cbase + (lane & 15);
    if (n0 + 32 < NKN) __builtin_prefetch(bp + 32 * CD, 0, 1);
    v16h b[8];
#pragma unroll
    for (int t = 0; t < 8; ++t) b[t] = load_b_frag(bp + t * 16, CD, lane);
#pragma unroll
    for (int t = 0; t < 8; ++t) acc2[t] = wmma_f16(a, b[t], acc2[t]);
  }

  // --- store out[q][k][c] ---
#pragma unroll
  for (int t = 0; t < 8; ++t) {
    int c = cbase + t * 16 + nn;
#pragma unroll
    for (int r = 0; r < 8; ++r) {
      int q = q0 + rt * 16 + r + 8 * h;
      out[((size_t)q * BKN + k) * CD + c] = (_Float16)acc2[t][r];
    }
  }
}

// ---------------------------------------------------------------------------
// Host launcher.  Inputs (setup_inputs order):
//  0:qx f16[256*512] 1:kx f16[256*512*512] 2:mask bool[256*512]
//  3:ln_q_g 4:ln_q_b 5:ln_k_g 6:ln_k_b f16[512]  7:wq 8:wk f16[512*512]
// Workspace: wqT | wkT | Q | KT
// ---------------------------------------------------------------------------
extern "C" void kernel_launch(void* const* d_in, const int* in_sizes, int n_in,
                              void* d_out, int out_size, void* d_ws, size_t ws_size,
                              hipStream_t stream) {
  (void)in_sizes; (void)n_in; (void)out_size; (void)ws_size;
  const _Float16* qx   = (const _Float16*)d_in[0];
  const _Float16* kx   = (const _Float16*)d_in[1];
  const unsigned char* mask = (const unsigned char*)d_in[2];
  const _Float16* lnqg = (const _Float16*)d_in[3];
  const _Float16* lnqb = (const _Float16*)d_in[4];
  const _Float16* lnkg = (const _Float16*)d_in[5];
  const _Float16* lnkb = (const _Float16*)d_in[6];
  const _Float16* wq   = (const _Float16*)d_in[7];
  const _Float16* wk   = (const _Float16*)d_in[8];
  _Float16* out = (_Float16*)d_out;

  _Float16* wqT = (_Float16*)d_ws;
  _Float16* wkT = wqT + (size_t)CD * CD;
  _Float16* Qp  = wkT + (size_t)CD * CD;
  _Float16* KT  = Qp  + (size_t)BQN * CD;

  const size_t ldsA = 32 * LDS_LD * sizeof(_Float16);            // 33280 B
  const size_t ldsB = ldsA + 32 * 8 * sizeof(float);             // + red

  transpose512_kernel<<<1024, 256, 0, stream>>>(wq, wqT);
  transpose512_kernel<<<1024, 256, 0, stream>>>(wk, wkT);
  // Q projection: 256 rows -> 8 blocks
  ln_proj_kernel<<<8, 256, ldsA, stream>>>(qx, lnqg, lnqb, wqT, Qp, 0);
  // K projection: 131072 rows -> 4096 blocks, stored batch-transposed
  ln_proj_kernel<<<4096, 256, ldsA, stream>>>(kx, lnkg, lnkb, wkT, KT, 1);
  // attention: (k, q-tile of 32 rows)
  attn_kernel<<<dim3(BKN, BQN / 32), 256, ldsB, stream>>>(Qp, KT, kx, mask, out);
}
